// MiniARDepthTransformer_75505525064725
// MI455X (gfx1250) — matrix operand order repaired and checked by reading, not compiled, not run on hardware
//
#include <hip/hip_runtime.h>
#include <hip/hip_bf16.h>

typedef __bf16 bf16_t;
typedef __bf16 v16bf __attribute__((ext_vector_type(16)));
typedef float  v8f   __attribute__((ext_vector_type(8)));

union Frag16 { uint4 u[2]; v16bf v; };

static constexpr int EPI_F32  = 0;  // out f32 = acc (+bias)
static constexpr int EPI_BF16 = 1;  // out bf16 = acc (+bias)
static constexpr int EPI_ADD  = 2;  // out f32 += acc (residual)

// C[M,N] = A[M,K](bf16) @ W[K,N](f32 -> bf16 in LDS)
// block = 256 threads = 8 waves; block tile 128x128; wave tile 16x128.
// Software-pipelined: next W stage (64x128) is prefetched into registers while
// the current stage's 16 WMMAs run; B fragments double-buffered across ds_loads.
template<int EPI, int K, int N, bool HAS_BIAS>
__global__ __launch_bounds__(256)
void gemm_wmma_bf16(const bf16_t* __restrict__ A,
                    const float*  __restrict__ W,
                    const float*  __restrict__ bias,
                    void* __restrict__ outv,
                    int M)
{
  constexpr int WP = 136;                  // LDS row pitch in halfs (272B, bank-spread)
  __shared__ bf16_t Wlds[64 * WP];

  const int tid   = threadIdx.x;
  const int lane  = tid & 31;              // wave32 (gfx1250)
  const int wave  = tid >> 5;
  const int nlane = lane & 15;
  const int khalf = lane >> 4;

  const int n0   = blockIdx.x * 128;
  const int row0 = blockIdx.y * 128 + wave * 16;

  v8f acc[8] = {};

  // ISA 16-bit A 16x32 layout: lanes 0-15 M=lane, VGPR0-3 K=khalf*8+0..7,
  // VGPR4-7 K=16+khalf*8+0..7 (lanes 16-31 take the khalf=1 halves).
  const bf16_t* Arow = A + (size_t)(row0 + nlane) * K;

  const int wr = tid >> 2;                 // 0..63 : K row of W stage
  const int wc = (tid & 3) * 32;           // 32-float column segment
  const float* Wtb = W + (size_t)wr * N + n0 + wc;
  bf16_t* Wdst = &Wlds[wr * WP + wc];

  // prologue: prefetch stage 0 of W into registers
  float4 wreg[8];
  #pragma unroll
  for (int i = 0; i < 8; ++i) wreg[i] = *(const float4*)(Wtb + i * 4);

  for (int k0 = 0; k0 < K; k0 += 64) {
    __syncthreads();                       // all waves done reading previous stage
    { // convert + store prefetched W stage (64 x 128) to LDS
      #pragma unroll
      for (int i = 0; i < 8; ++i) {
        Wdst[i * 4 + 0] = (bf16_t)wreg[i].x;
        Wdst[i * 4 + 1] = (bf16_t)wreg[i].y;
        Wdst[i * 4 + 2] = (bf16_t)wreg[i].z;
        Wdst[i * 4 + 3] = (bf16_t)wreg[i].w;
      }
    }
    __syncthreads();

    // prefetch next W stage into registers (overlaps with the 16 WMMAs below)
    if (k0 + 64 < K) {
      const float* wp = Wtb + (size_t)(k0 + 64) * N;
      #pragma unroll
      for (int i = 0; i < 8; ++i) wreg[i] = *(const float4*)(wp + i * 4);
    }

    // both K=32 A sub-fragments up front
    Frag16 a[2];
    #pragma unroll
    for (int s = 0; s < 2; ++s) {
      const bf16_t* ap = Arow + k0 + s * 32;
      a[s].u[0] = *(const uint4*)(ap + khalf * 8);
      a[s].u[1] = *(const uint4*)(ap + 16 + khalf * 8);
    }

    // 16 WMMAs with double-buffered B fragments
    // ISA 16-bit B 32x16 layout: N column = nlane, halfs 0-7 K=khalf*8+0..7, 8-15 K=16+khalf*8+0..7
    Frag16 b[2];
    {
      #pragma unroll
      for (int e = 0; e < 16; ++e) {
        const int kk = (e < 8) ? (khalf * 8 + e) : (16 + khalf * 8 + (e - 8));
        b[0].v[e] = Wlds[kk * WP + nlane];
      }
    }
    #pragma unroll
    for (int t = 0; t < 16; ++t) {         // t = s*8 + nt
      if (t < 15) {
        const int tn = t + 1;
        #pragma unroll
        for (int e = 0; e < 16; ++e) {
          const int kk = (e < 8) ? (khalf * 8 + e) : (16 + khalf * 8 + (e - 8));
          b[tn & 1].v[e] = Wlds[(((tn >> 3) * 32) + kk) * WP + (tn & 7) * 16 + nlane];
        }
      }
      acc[t & 7] = __builtin_amdgcn_wmma_f32_16x16x32_bf16(
          false, a[t >> 3].v, false, b[t & 1].v, (short)0, acc[t & 7], false, false);
    }
  }

  // Epilogue. C layout: VGPR g -> M = g + 8*khalf, N-in-tile = nlane.
  float bv[8];
  #pragma unroll
  for (int nt = 0; nt < 8; ++nt)
    bv[nt] = HAS_BIAS ? bias[n0 + nt * 16 + nlane] : 0.0f;

  const size_t base = (size_t)(row0 + khalf * 8) * N + n0 + nlane;
  if (EPI == EPI_ADD) {
    float* out = (float*)outv + base;
    #pragma unroll
    for (int g = 0; g < 8; ++g)
      #pragma unroll
      for (int nt = 0; nt < 8; ++nt)
        out[(size_t)g * N + nt * 16] += acc[nt][g];
  } else if (EPI == EPI_F32) {
    float* out = (float*)outv + base;
    #pragma unroll
    for (int g = 0; g < 8; ++g)
      #pragma unroll
      for (int nt = 0; nt < 8; ++nt)
        out[(size_t)g * N + nt * 16] = acc[nt][g] + bv[nt];
  } else {
    bf16_t* out = (bf16_t*)outv + base;
    #pragma unroll
    for (int g = 0; g < 8; ++g)
      #pragma unroll
      for (int nt = 0; nt < 8; ++nt)
        out[(size_t)g * N + nt * 16] = (bf16_t)(acc[nt][g] + bv[nt]);
  }
}

// out_bf16[row,:] = x[row,:] * rsqrt(mean(x^2)+eps) * w ; one wave per 512-wide row
__global__ __launch_bounds__(256)
void rmsnorm_to_bf16(const float* __restrict__ x, const float* __restrict__ w,
                     bf16_t* __restrict__ out, int M)
{
  const int lane = threadIdx.x & 31;
  const int row  = blockIdx.x * 8 + (threadIdx.x >> 5);
  if (row >= M) return;
  const float* xr = x + (size_t)row * 512 + lane * 16;
  float v[16];
  #pragma unroll
  for (int i = 0; i < 16; i += 4) {
    float4 f = *(const float4*)(xr + i);
    v[i] = f.x; v[i + 1] = f.y; v[i + 2] = f.z; v[i + 3] = f.w;
  }
  float ss = 0.f;
  #pragma unroll
  for (int i = 0; i < 16; ++i) ss += v[i] * v[i];
  #pragma unroll
  for (int off = 16; off > 0; off >>= 1) ss += __shfl_xor(ss, off, 32);
  const float sc = rsqrtf(ss * (1.0f / 512.0f) + 1e-6f);
  const float* wr = w + lane * 16;
  union { bf16_t h[16]; uint4 u[2]; } o;
  #pragma unroll
  for (int i = 0; i < 16; ++i) o.h[i] = (bf16_t)(v[i] * sc * wr[i]);
  uint4* op = (uint4*)(out + (size_t)row * 512 + lane * 16);
  op[0] = o.u[0]; op[1] = o.u[1];
}

__global__ void embed_gather(const int* __restrict__ tok,
                             const float* __restrict__ emb,
                             float* __restrict__ x, int total)
{
  int i = blockIdx.x * 256 + threadIdx.x;
  if (i >= total) return;
  int b = i >> 9, d = i & 511;                 // D = 512
  x[i] = emb[(size_t)tok[b] * 512 + d];
}

__global__ void cvt_f32_bf16(const float* __restrict__ in,
                             bf16_t* __restrict__ out, int n)
{
  int i = blockIdx.x * 256 + threadIdx.x;
  if (i < n) out[i] = (bf16_t)in[i];
}

// g[i] = silu(g[i]) * u[i]  (in place on g)
__global__ void silu_mul_inplace(bf16_t* __restrict__ g,
                                 const bf16_t* __restrict__ u, int n)
{
  int i = blockIdx.x * 256 + threadIdx.x;
  if (i >= n) return;
  float gf = (float)g[i];
  float s  = gf / (1.0f + __expf(-gf));
  g[i] = (bf16_t)(s * (float)u[i]);
}

extern "C" void kernel_launch(void* const* d_in, const int* in_sizes, int n_in,
                              void* d_out, int out_size, void* d_ws, size_t ws_size,
                              hipStream_t stream)
{
  const int*   tok   = (const int*)  d_in[0];
  const float* mh    = (const float*)d_in[1];
  const float* emb   = (const float*)d_in[2];
  const float* Wc    = (const float*)d_in[3];
  const float* bc    = (const float*)d_in[4];
  const float* sa_n  = (const float*)d_in[5];
  // d_in[6], d_in[7] = sa_q, sa_k : dead (S=T=1 -> softmax == 1)
  const float* sa_v  = (const float*)d_in[8];
  const float* sa_o  = (const float*)d_in[9];
  // d_in[10..12] = ca_n, ca_q, ca_k : dead (ca output = (context@ca_v)@ca_o)
  const float* ca_v  = (const float*)d_in[13];
  const float* ca_o  = (const float*)d_in[14];
  const float* ffn_n = (const float*)d_in[15];
  const float* w_g   = (const float*)d_in[16];
  const float* w_u   = (const float*)d_in[17];
  const float* w_d   = (const float*)d_in[18];
  const float* fin_n = (const float*)d_in[19];
  const float* Wout  = (const float*)d_in[20];
  (void)n_in; (void)out_size; (void)ws_size;

  const int B = in_sizes[0];               // 8192 (multiple of 128)
  const int D = 512, F = 1024, MD = 768, V = 2048, L = 4;

  char* ws = (char*)d_ws; size_t off = 0;
  auto take = [&](size_t bytes) -> void* {
    void* p = ws + off; off = (off + bytes + 255) & ~(size_t)255; return p;
  };
  float*  x    = (float*) take((size_t)B * D * 4);
  bf16_t* hb   = (bf16_t*)take((size_t)B * D * 2);
  bf16_t* t1   = (bf16_t*)take((size_t)B * D * 2);
  bf16_t* mhb  = (bf16_t*)take((size_t)B * MD * 2);
  bf16_t* ctxb = (bf16_t*)take((size_t)B * D * 2);
  bf16_t* gb   = (bf16_t*)take((size_t)B * F * 2);
  bf16_t* ub   = (bf16_t*)take((size_t)B * F * 2);

  dim3 blk(256);
  dim3 gD (D / 128, B / 128);
  dim3 gF (F / 128, B / 128);
  dim3 gV (V / 128, B / 128);
  dim3 gRn((B + 7) / 8);

  // prep: x = emb[tokens]; mh -> bf16; context = mh @ Wc + bc (store bf16)
  cvt_f32_bf16<<<dim3((B * MD + 255) / 256), blk, 0, stream>>>(mh, mhb, B * MD);
  embed_gather<<<dim3((B * D + 255) / 256), blk, 0, stream>>>(tok, emb, x, B * D);
  gemm_wmma_bf16<EPI_BF16, 768, 512, true><<<gD, blk, 0, stream>>>(mhb, Wc, bc, ctxb, B);

  for (int i = 0; i < L; ++i) {
    const float* svi = sa_v + (size_t)i * D * D;
    const float* soi = sa_o + (size_t)i * D * D;
    const float* cvi = ca_v + (size_t)i * D * D;
    const float* coi = ca_o + (size_t)i * D * D;
    const float* wgi = w_g + (size_t)i * D * F;
    const float* wui = w_u + (size_t)i * D * F;
    const float* wdi = w_d + (size_t)i * F * D;

    // self-attn (degenerate): x += (rmsnorm(x)*sa_n @ sa_v) @ sa_o
    rmsnorm_to_bf16<<<gRn, blk, 0, stream>>>(x, sa_n + i * D, hb, B);
    gemm_wmma_bf16<EPI_BF16, 512, 512, false><<<gD, blk, 0, stream>>>(hb, svi, nullptr, t1, B);
    gemm_wmma_bf16<EPI_ADD,  512, 512, false><<<gD, blk, 0, stream>>>(t1, soi, nullptr, x,  B);

    // cross-attn (degenerate): x += (context @ ca_v) @ ca_o
    gemm_wmma_bf16<EPI_BF16, 512, 512, false><<<gD, blk, 0, stream>>>(ctxb, cvi, nullptr, t1, B);
    gemm_wmma_bf16<EPI_ADD,  512, 512, false><<<gD, blk, 0, stream>>>(t1, coi, nullptr, x,  B);

    // SwiGLU FFN: x += (silu(h@w_g) * (h@w_u)) @ w_d
    rmsnorm_to_bf16<<<gRn, blk, 0, stream>>>(x, ffn_n + i * D, hb, B);
    gemm_wmma_bf16<EPI_BF16, 512, 1024, false><<<gF, blk, 0, stream>>>(hb, wgi, nullptr, gb, B);
    gemm_wmma_bf16<EPI_BF16, 512, 1024, false><<<gF, blk, 0, stream>>>(hb, wui, nullptr, ub, B);
    silu_mul_inplace<<<dim3((B * F + 255) / 256), blk, 0, stream>>>(gb, ub, B * F);
    gemm_wmma_bf16<EPI_ADD, 1024, 512, false><<<gD, blk, 0, stream>>>(gb, wdi, nullptr, x, B);
  }

  // final: out = rmsnorm(x, final_n) @ Wout  (f32 store to d_out)
  rmsnorm_to_bf16<<<gRn, blk, 0, stream>>>(x, fin_n, hb, B);
  gemm_wmma_bf16<EPI_F32, 512, 2048, false><<<gV, blk, 0, stream>>>(hb, Wout, nullptr, d_out, B);
}
